// PointNeXT_14912126451978
// MI455X (gfx1250) — matrix operand order repaired and checked by reading, not compiled, not run on hardware
//
#include <hip/hip_runtime.h>
#include <hip/hip_bf16.h>
#include <math.h>

typedef __attribute__((ext_vector_type(16))) _Float16 v16h;
typedef __attribute__((ext_vector_type(8)))  _Float16 v8h;
typedef __attribute__((ext_vector_type(8)))  float    v8f;

static __host__ __device__ inline int ceil_div(int a, int b) { return (a + b - 1) / b; }
static __host__ __device__ inline int pad32(int c) { return (c + 31) & ~31; }

// ---------------------------------------------------------------------------
// WMMA fused linear on padded-f16 operands, TM x TN register-tiled:
//   X : (rows, cinP)  f16, cinP % 32 == 0, pads zero;  rows % (TM*16) == 0
//   Wh: (coutP, cinP) f16, zero-padded rows/cols;      coutP % (TN*16) == 0
//   Y : (rows, coutP) f16; pad columns written as zero
//   flags bit0 = relu, bit1 = batchnorm affine
// One wave per (TM*16 x TN*16) super-tile; __launch_bounds__(32) so the
// register allocator gets the full VGPR file (no accumulator spills).
// K-loop is software-pipelined: next fragments are in flight during wmmas.
// Epilogue transposes tiles through LDS for coalesced b128 row stores.
// ---------------------------------------------------------------------------
template <int TM, int TN>
__global__ __launch_bounds__(32)
void wmma_linear_kernel(const _Float16* __restrict__ X,
                        const _Float16* __restrict__ Wh,
                        const float* __restrict__ bias,
                        const float* __restrict__ beta,
                        const float* __restrict__ g,
                        _Float16* __restrict__ Y,
                        int rows, int cinP, int cout, int coutP,
                        int flags) {
  __shared__ __align__(16) _Float16 tile[TM * TN * 256];
  const int lane = threadIdx.x;          // blockDim.x == 32 (one wave)
  const int half = lane >> 4;
  const int lm   = lane & 15;
  const int row0 = blockIdx.x * (TM * 16);
  const int col0 = blockIdx.y * (TN * 16);

  const _Float16* xrow[TM];
#pragma unroll
  for (int t = 0; t < TM; ++t) xrow[t] = X + (size_t)(row0 + t * 16 + lm) * cinP;
  const _Float16* wrow[TN];
#pragma unroll
  for (int t = 0; t < TN; ++t) wrow[t] = Wh + (size_t)(col0 + t * 16 + lm) * cinP;

  // A 16x32 f16: lanes 0-15 hold K 0..15; lanes 16-31 hold K 16..31
  auto loadA = [&](int t, int k0) -> v16h {
    return *(const v16h*)(xrow[t] + k0 + half * 16);
  };
  // B 32x16 f16: lanes 0-15 hold K 0..15; lanes 16-31 hold K 16..31
  auto loadB = [&](int t, int k0) -> v16h {
    return *(const v16h*)(wrow[t] + k0 + half * 16);
  };

  v8f acc[TM][TN] = {};
  v16h aCur[TM], bCur[TN];
#pragma unroll
  for (int t = 0; t < TM; ++t) aCur[t] = loadA(t, 0);
#pragma unroll
  for (int t = 0; t < TN; ++t) bCur[t] = loadB(t, 0);

  for (int k0 = 0; k0 < cinP; k0 += 32) {
    const int kn = k0 + 32;
    v16h aNxt[TM], bNxt[TN];
    if (kn < cinP) {                     // uniform branch: preload next K-step
#pragma unroll
      for (int t = 0; t < TM; ++t) {
        __builtin_prefetch(xrow[t] + kn + 32, 0, 3);
        aNxt[t] = loadA(t, kn);
      }
#pragma unroll
      for (int t = 0; t < TN; ++t) bNxt[t] = loadB(t, kn);
    }
#pragma unroll
    for (int j = 0; j < TN; ++j)
#pragma unroll
      for (int i = 0; i < TM; ++i)
        acc[i][j] = __builtin_amdgcn_wmma_f32_16x16x32_f16(false, aCur[i], false, bCur[j],
                                                           (short)0, acc[i][j], false, false);
    if (kn < cinP) {
#pragma unroll
      for (int t = 0; t < TM; ++t) aCur[t] = aNxt[t];
#pragma unroll
      for (int t = 0; t < TN; ++t) bCur[t] = bNxt[t];
    }
  }

  // epilogue: D element q of tile (i,j) lives at (m = q + 8*half, n = lane%16)
#pragma unroll
  for (int j = 0; j < TN; ++j) {
    const int cB = col0 + j * 16 + lm;
    const bool colOK = (cB < cout);
    float bn_s = 1.0f, bn_b = 0.0f, bv = 0.0f;
    if (colOK) {
      bv = bias[cB];
      if (flags & 2) { bn_s = g[cB] * rsqrtf(1.0f + 1e-5f); bn_b = beta[cB]; }
    }
#pragma unroll
    for (int i = 0; i < TM; ++i) {
      _Float16* tl = tile + (i * TN + j) * 256;
#pragma unroll
      for (int q = 0; q < 8; ++q) {
        float v = colOK ? (bn_s * (acc[i][j][q] + bv) + bn_b) : 0.0f;
        if (flags & 1) v = fmaxf(v, 0.0f);
        tl[(q + 8 * half) * 16 + lm] = (_Float16)v;
      }
    }
  }
  __syncthreads();
  // coalesced stores: lane -> (row r = lane/2, 8-wide chunk hc = lane&1)
  const int r = lane >> 1, hc = lane & 1;
#pragma unroll
  for (int i = 0; i < TM; ++i) {
    const int gm = row0 + i * 16 + r;
#pragma unroll
    for (int j = 0; j < TN; ++j) {
      v8h vv = *(const v8h*)(tile + (i * TN + j) * 256 + r * 16 + hc * 8);
      *(v8h*)(Y + (size_t)gm * coutP + col0 + j * 16 + hc * 8) = vv;
    }
  }
}

// f32 (rows x cin) -> f16 padded (rowsP x cinP), zero-filled pad
__global__ void cvt_pad_kernel(const float* __restrict__ in, _Float16* __restrict__ out,
                               int rows, int cin, int cinP, int rowsP) {
  int i = blockIdx.x * blockDim.x + threadIdx.x;
  if (i >= rowsP * cinP) return;
  int r = i / cinP, c = i % cinP;
  float v = (r < rows && c < cin) ? in[(size_t)r * cin + c] : 0.0f;
  out[i] = (_Float16)v;
}

// (B,3,N) -> (B,N,3) f32 geometry
__global__ void transpose_xyz_kernel(const float* __restrict__ in, float* __restrict__ out,
                                     int B, int N) {
  int i = blockIdx.x * blockDim.x + threadIdx.x;
  if (i >= B * N) return;
  int b = i / N, n = i % N;
  out[(size_t)i * 3 + 0] = in[((size_t)b * 3 + 0) * N + n];
  out[(size_t)i * 3 + 1] = in[((size_t)b * 3 + 1) * N + n];
  out[(size_t)i * 3 + 2] = in[((size_t)b * 3 + 2) * N + n];
}

// Farthest point sampling: one block per batch, LDS argmax reduction.
__global__ void fps_kernel(const float* __restrict__ xyz, int* __restrict__ fidx,
                           int N, int npoint) {
  const int b   = blockIdx.x;
  const int tid = threadIdx.x;           // blockDim.x == 256
  const float* p = xyz + (size_t)b * N * 3;
  __shared__ float sdist[256];
  __shared__ int   sidx[256];
  __shared__ int   s_far;
  float mind[16];
  const int per = ceil_div(N, 256);
#pragma unroll
  for (int j = 0; j < 16; ++j) mind[j] = 1e10f;
  if (tid == 0) s_far = 0;
  __syncthreads();
  for (int it = 0; it < npoint; ++it) {
    const int far = s_far;
    if (tid == 0) fidx[b * npoint + it] = far;
    const float cx = p[far * 3 + 0], cy = p[far * 3 + 1], cz = p[far * 3 + 2];
    float best = -1.0f; int bi = 0;
    for (int j = 0; j < per; ++j) {
      const int i = tid + j * 256;
      if (i < N) {
        float dx = p[i * 3 + 0] - cx, dy = p[i * 3 + 1] - cy, dz = p[i * 3 + 2] - cz;
        float d = dx * dx + dy * dy + dz * dz;
        float m = fminf(mind[j], d);
        mind[j] = m;
        if (m > best) { best = m; bi = i; }
      }
    }
    sdist[tid] = best; sidx[tid] = bi;
    __syncthreads();
    for (int s = 128; s > 0; s >>= 1) {
      if (tid < s && sdist[tid + s] > sdist[tid]) {
        sdist[tid] = sdist[tid + s]; sidx[tid] = sidx[tid + s];
      }
      __syncthreads();
    }
    if (tid == 0) s_far = sidx[0];
    __syncthreads();
  }
}

__global__ void gather_xyz_kernel(const float* __restrict__ xyz, const int* __restrict__ fidx,
                                  float* __restrict__ out, int B, int N, int M) {
  int i = blockIdx.x * blockDim.x + threadIdx.x;
  if (i >= B * M) return;
  int b = i / M; int id = fidx[i];
  const float* s = xyz + ((size_t)b * N + id) * 3;
  out[(size_t)i * 3 + 0] = s[0];
  out[(size_t)i * 3 + 1] = s[1];
  out[(size_t)i * 3 + 2] = s[2];
}

__global__ void ball_query_kernel(const float* __restrict__ xyz, const float* __restrict__ new_xyz,
                                  int* __restrict__ idx, int B, int N, int M, int ns, float r2) {
  int i = blockIdx.x * blockDim.x + threadIdx.x;
  if (i >= B * M) return;
  int b = i / M;
  const float* q = new_xyz + (size_t)i * 3;
  const float* p = xyz + (size_t)b * N * 3;
  const float qx = q[0], qy = q[1], qz = q[2];
  int cnt = 0, first = 0;
  for (int j = 0; j < N && cnt < ns; ++j) {
    float dx = p[j * 3 + 0] - qx, dy = p[j * 3 + 1] - qy, dz = p[j * 3 + 2] - qz;
    if (dx * dx + dy * dy + dz * dz <= r2) {
      if (cnt == 0) first = j;
      idx[(size_t)i * ns + cnt] = j;
      ++cnt;
    }
  }
  for (; cnt < ns; ++cnt) idx[(size_t)i * ns + cnt] = first;
}

// feat[b,m,s,:] = concat(xyz[idx]-new_xyz[m], points[idx]) -> f16 (B*M*ns, CfP)
__global__ void group_points_kernel(const float* __restrict__ xyz, const float* __restrict__ new_xyz,
                                    const _Float16* __restrict__ points, const int* __restrict__ idx,
                                    _Float16* __restrict__ feat, int B, int N, int M, int ns,
                                    int C, int CpIn, int CfP) {
  int i = blockIdx.x * blockDim.x + threadIdx.x;
  const int total = B * M * ns;
  if (i >= total) return;
  int b = i / (M * ns);
  int rem = i % (M * ns);
  int m = rem / ns;
  int id = idx[i];
  const float* s  = xyz + ((size_t)b * N + id) * 3;
  const float* nc = new_xyz + ((size_t)b * M + m) * 3;
  _Float16* o = feat + (size_t)i * CfP;
  o[0] = (_Float16)(s[0] - nc[0]);
  o[1] = (_Float16)(s[1] - nc[1]);
  o[2] = (_Float16)(s[2] - nc[2]);
  const _Float16* ps = points + ((size_t)b * N + id) * CpIn;
  for (int c = 0; c < C; ++c) o[3 + c] = ps[c];
  for (int c = 3 + C; c < CfP; ++c) o[c] = (_Float16)0.0f;  // keep pads zero
}

// (BM, ns, Cp) -> (BM, Cp) max over ns (pads are zero, max stays zero)
__global__ void maxpool_kernel(const _Float16* __restrict__ in, _Float16* __restrict__ out,
                               int BM, int ns, int Cp) {
  int i = blockIdx.x * blockDim.x + threadIdx.x;
  if (i >= BM * Cp) return;
  int bm = i / Cp, c = i % Cp;
  const _Float16* p = in + (size_t)bm * ns * Cp + c;
  float m = -1e30f;
  for (int s0 = 0; s0 < ns; ++s0) m = fmaxf(m, (float)p[(size_t)s0 * Cp]);
  out[i] = (_Float16)m;
}

__global__ void add_relu_kernel(const _Float16* __restrict__ a, const _Float16* __restrict__ b,
                                _Float16* __restrict__ out, int n) {
  int i = blockIdx.x * blockDim.x + threadIdx.x;
  if (i < n) out[i] = (_Float16)fmaxf((float)a[i] + (float)b[i], 0.0f);
}

__global__ void three_nn_kernel(const float* __restrict__ xyz1, const float* __restrict__ xyz2,
                                int* __restrict__ idx3, float* __restrict__ w3,
                                int B, int n1, int n2) {
  int i = blockIdx.x * blockDim.x + threadIdx.x;
  if (i >= B * n1) return;
  int b = i / n1;
  const float* q = xyz1 + (size_t)i * 3;
  const float* p = xyz2 + (size_t)b * n2 * 3;
  const float qx = q[0], qy = q[1], qz = q[2];
  float d0 = 1e30f, d1 = 1e30f, d2 = 1e30f; int i0 = 0, i1 = 0, i2 = 0;
  for (int j = 0; j < n2; ++j) {
    float dx = p[j * 3 + 0] - qx, dy = p[j * 3 + 1] - qy, dz = p[j * 3 + 2] - qz;
    float d = dx * dx + dy * dy + dz * dz;
    if (d < d0)      { d2 = d1; i2 = i1; d1 = d0; i1 = i0; d0 = d; i0 = j; }
    else if (d < d1) { d2 = d1; i2 = i1; d1 = d;  i1 = j; }
    else if (d < d2) { d2 = d;  i2 = j; }
  }
  float w0 = 1.0f / (d0 + 1e-8f), w1 = 1.0f / (d1 + 1e-8f), w2 = 1.0f / (d2 + 1e-8f);
  float ws = w0 + w1 + w2;
  idx3[i * 3 + 0] = i0; idx3[i * 3 + 1] = i1; idx3[i * 3 + 2] = i2;
  w3[i * 3 + 0] = w0 / ws; w3[i * 3 + 1] = w1 / ws; w3[i * 3 + 2] = w2 / ws;
}

// interp into strided f16 concat buffer: out[pt*CtP + c0 + c]
__global__ void three_interp_kernel(const _Float16* __restrict__ pts2, const int* __restrict__ idx3,
                                    const float* __restrict__ w3, _Float16* __restrict__ out,
                                    int B, int n1, int n2, int C2, int C2p, int CtP, int c0) {
  int i = blockIdx.x * blockDim.x + threadIdx.x;
  if (i >= B * n1 * C2) return;
  int c = i % C2; int pt = i / C2; int b = pt / n1;
  const _Float16* base = pts2 + (size_t)b * n2 * C2p;
  float v = 0.0f;
#pragma unroll
  for (int j = 0; j < 3; ++j)
    v += w3[pt * 3 + j] * (float)base[(size_t)idx3[pt * 3 + j] * C2p + c];
  out[(size_t)pt * CtP + c0 + c] = (_Float16)v;
}

__global__ void copy_channels_kernel(const _Float16* __restrict__ in, _Float16* __restrict__ out,
                                     int rows, int C1, int C1p, int CtP, int off) {
  int i = blockIdx.x * blockDim.x + threadIdx.x;
  if (i >= rows * C1) return;
  int r = i / C1, c = i % C1;
  out[(size_t)r * CtP + off + c] = in[(size_t)r * C1p + c];
}

// f16 (B,N,Cp) -> log_softmax over first C, transposed f32 (B,C,N)
__global__ void head_logsoftmax_kernel(const _Float16* __restrict__ in, float* __restrict__ out,
                                       int B, int N, int C, int Cp) {
  int i = blockIdx.x * blockDim.x + threadIdx.x;
  if (i >= B * N) return;
  int b = i / N, n = i % N;
  const _Float16* x = in + (size_t)i * Cp;
  float mx = -1e30f;
  for (int c = 0; c < C; ++c) mx = fmaxf(mx, (float)x[c]);
  float sum = 0.0f;
  for (int c = 0; c < C; ++c) sum += __expf((float)x[c] - mx);
  float lse = mx + __logf(sum);
  for (int c = 0; c < C; ++c)
    out[((size_t)b * C + c) * N + n] = (float)x[c] - lse;
}

// ---------------------------------------------------------------------------
// Host-side orchestration
// ---------------------------------------------------------------------------
struct LayerP { const float *W, *b, *beta, *g; };
struct Scratch {
  int* fidx; int* idx; int* idx3; float* w3;
  _Float16* bufA; _Float16* bufB; _Float16* bufC; _Float16* bufW;
};

static inline void launch_linear(const _Float16* X, const LayerP& L, _Float16* Y,
                                 int rows, int cin, int cout, int flags,
                                 Scratch& sc, hipStream_t s) {
  const int cinP = pad32(cin), coutP = pad32(cout);
  cvt_pad_kernel<<<ceil_div(coutP * cinP, 256), 256, 0, s>>>(L.W, sc.bufW, cout, cin, cinP, coutP);
  if (coutP % 64 == 0) {
    dim3 grid(rows / 32, coutP / 64);
    wmma_linear_kernel<2, 4><<<grid, 32, 0, s>>>(X, sc.bufW, L.b, L.beta, L.g, Y,
                                                 rows, cinP, cout, coutP, flags);
  } else {
    dim3 grid(rows / 32, coutP / 32);
    wmma_linear_kernel<2, 2><<<grid, 32, 0, s>>>(X, sc.bufW, L.b, L.beta, L.g, Y,
                                                 rows, cinP, cout, coutP, flags);
  }
}

static void run_sa(const float* xyz, const _Float16* points, int B, int N, int C, int CpIn,
                   int npoint, float radius, int ns,
                   const LayerP* L, const int* dims, int nl,
                   float* out_xyz, _Float16* out_pts, Scratch& sc, hipStream_t s) {
  fps_kernel<<<B, 256, 0, s>>>(xyz, sc.fidx, N, npoint);
  gather_xyz_kernel<<<ceil_div(B * npoint, 256), 256, 0, s>>>(xyz, sc.fidx, out_xyz, B, N, npoint);
  ball_query_kernel<<<ceil_div(B * npoint, 128), 128, 0, s>>>(xyz, out_xyz, sc.idx, B, N, npoint, ns,
                                                              radius * radius);
  const int rows = B * npoint * ns;
  group_points_kernel<<<ceil_div(rows, 256), 256, 0, s>>>(xyz, out_xyz, points, sc.idx, sc.bufA,
                                                          B, N, npoint, ns, C, CpIn, pad32(dims[0]));
  const _Float16* cur = sc.bufA;
  _Float16* ping[2] = { sc.bufB, sc.bufA };
  int t = 0;
  for (int i = 0; i < nl; ++i) {
    _Float16* dst = ping[t]; t ^= 1;
    launch_linear(cur, L[i], dst, rows, dims[i], dims[i + 1], 3, sc, s);
    cur = dst;
  }
  const int Cp = pad32(dims[nl]);
  maxpool_kernel<<<ceil_div(B * npoint * Cp, 256), 256, 0, s>>>(cur, out_pts, B * npoint, ns, Cp);
}

static void run_invres(const float* xyz, _Float16* points, int B, int N, int C,
                       float radius, int ns, const LayerP* L, const int* dims,
                       Scratch& sc, hipStream_t s) {
  ball_query_kernel<<<ceil_div(B * N, 128), 128, 0, s>>>(xyz, xyz, sc.idx, B, N, N, ns,
                                                         radius * radius);
  const int rows = B * N * ns;
  group_points_kernel<<<ceil_div(rows, 256), 256, 0, s>>>(xyz, xyz, points, sc.idx, sc.bufA,
                                                          B, N, N, ns, C, C, pad32(dims[0]));
  launch_linear(sc.bufA, L[0], sc.bufB, rows, dims[0], dims[1], 3, sc, s);
  const int h0p = pad32(dims[1]);
  maxpool_kernel<<<ceil_div(B * N * h0p, 256), 256, 0, s>>>(sc.bufB, sc.bufC, B * N, ns, h0p);
  launch_linear(sc.bufC, L[1], sc.bufA, B * N, dims[1], dims[2], 3, sc, s);
  launch_linear(sc.bufA, L[2], sc.bufB, B * N, dims[2], dims[3], 2, sc, s);   // BN, no relu
  add_relu_kernel<<<ceil_div(B * N * C, 256), 256, 0, s>>>(sc.bufB, points, points, B * N * C);
}

static void run_fp(const float* xyz1, const float* xyz2, const _Float16* pts1, int C1,
                   const _Float16* pts2, int C2, int B, int n1, int n2,
                   const LayerP* L, const int* dims, int nl, _Float16* out,
                   Scratch& sc, hipStream_t s) {
  three_nn_kernel<<<ceil_div(B * n1, 128), 128, 0, s>>>(xyz1, xyz2, sc.idx3, sc.w3, B, n1, n2);
  const int Ct = C1 + C2;          // always a multiple of 32 in this net
  if (C1 > 0)
    copy_channels_kernel<<<ceil_div(B * n1 * C1, 256), 256, 0, s>>>(pts1, sc.bufA, B * n1, C1, C1, Ct, 0);
  three_interp_kernel<<<ceil_div(B * n1 * C2, 256), 256, 0, s>>>(pts2, sc.idx3, sc.w3, sc.bufA,
                                                                 B, n1, n2, C2, C2, Ct, C1);
  const _Float16* cur = sc.bufA;
  _Float16* ping[2] = { sc.bufB, sc.bufA };
  int t = 0;
  for (int i = 0; i < nl; ++i) {
    _Float16* dst = (i == nl - 1) ? out : ping[t]; t ^= 1;
    launch_linear(cur, L[i], dst, B * n1, dims[i], dims[i + 1], 3, sc, s);
    cur = dst;
  }
}

extern "C" void kernel_launch(void* const* d_in, const int* in_sizes, int n_in,
                              void* d_out, int out_size, void* d_ws, size_t ws_size,
                              hipStream_t stream) {
  (void)in_sizes; (void)n_in; (void)out_size; (void)ws_size;

  // JAX pytree order: sorted dict keys; per-layer dict sorts as W, b, beta, g;
  // top-level 'params' < 'xyz'.
  int ip = 0;
  auto nextp = [&]() { return (const float*)d_in[ip++]; };
  auto grab  = [&](LayerP* dst, int n) {
    for (int i = 0; i < n; ++i) { dst[i].W = nextp(); dst[i].b = nextp(); dst[i].beta = nextp(); dst[i].g = nextp(); }
  };
  LayerP fp1[3], fp2[2], fp3[2], fp4[2], head1[1];
  LayerP ir1[3], ir2[3], ir3[3], ir4[3], sa1[3], sa2[3], sa3[3], sa4[3];
  grab(fp1, 3); grab(fp2, 2); grab(fp3, 2); grab(fp4, 2); grab(head1, 1);
  LayerP head2; head2.W = nextp(); head2.b = nextp(); head2.beta = nullptr; head2.g = nullptr;
  grab(ir1, 3); grab(ir2, 3); grab(ir3, 3); grab(ir4, 3);
  grab(sa1, 3); grab(sa2, 3); grab(sa3, 3); grab(sa4, 3);
  const float* xyz_in = nextp();                    // (8, 3, 4096)

  // workspace bump allocator
  char* w = (char*)d_ws;
  size_t off = 0;
  auto alloc_f = [&](size_t n) { float* p = (float*)(w + off); off = (off + n * 4 + 255) & ~(size_t)255; return p; };
  auto alloc_i = [&](size_t n) { int*   p = (int*)(w + off);   off = (off + n * 4 + 255) & ~(size_t)255; return p; };
  auto alloc_h = [&](size_t n) { _Float16* p = (_Float16*)(w + off); off = (off + n * 2 + 255) & ~(size_t)255; return p; };

  const int B = 8, N0 = 4096;
  float*     pts    = alloc_f((size_t)B * N0 * 3);
  _Float16*  l0h    = alloc_h((size_t)B * N0 * 32);      // l0 points as f16 (3 real ch, stride 32)
  float*     l1_xyz = alloc_f((size_t)B * 1024 * 3);
  _Float16*  l1_pts = alloc_h((size_t)B * 1024 * 64);
  float*     l2_xyz = alloc_f((size_t)B * 256 * 3);
  _Float16*  l2_pts = alloc_h((size_t)B * 256 * 128);
  float*     l3_xyz = alloc_f((size_t)B * 64 * 3);
  _Float16*  l3_pts = alloc_h((size_t)B * 64 * 256);
  float*     l4_xyz = alloc_f((size_t)B * 16 * 3);
  _Float16*  l4_pts = alloc_h((size_t)B * 16 * 512);
  _Float16*  l3p    = alloc_h((size_t)B * 64 * 256);
  _Float16*  l2p    = alloc_h((size_t)B * 256 * 256);
  _Float16*  l1p    = alloc_h((size_t)B * 1024 * 128);
  _Float16*  l0p    = alloc_h((size_t)B * N0 * 128);

  Scratch sc;
  sc.fidx = alloc_i((size_t)B * 1024);
  sc.idx  = alloc_i((size_t)B * 1024 * 32);
  sc.idx3 = alloc_i((size_t)B * N0 * 3);
  sc.w3   = alloc_f((size_t)B * N0 * 3);
  sc.bufW = alloc_h((size_t)1024 * 544);                 // largest padded weight (ir4)
  sc.bufC = alloc_h((size_t)1 << 20);                    // pooled invres features
  const size_t CAP_H = (size_t)262144 * 96;              // largest grouped feature (ir1: 262144 x 96)
  sc.bufA = alloc_h(CAP_H);
  sc.bufB = alloc_h(CAP_H);

  const int d_sa1[4] = { 6, 32, 32, 64 };
  const int d_ir1[4] = { 67, 64, 256, 64 };
  const int d_sa2[4] = { 67, 64, 64, 128 };
  const int d_ir2[4] = { 131, 128, 512, 128 };
  const int d_sa3[4] = { 131, 128, 128, 256 };
  const int d_ir3[4] = { 259, 256, 1024, 256 };
  const int d_sa4[4] = { 259, 256, 256, 512 };
  const int d_ir4[4] = { 515, 512, 1024, 512 };
  const int d_fp4[3] = { 768, 256, 256 };
  const int d_fp3[3] = { 384, 256, 256 };
  const int d_fp2[3] = { 320, 256, 128 };
  const int d_fp1[4] = { 128, 128, 128, 128 };

  // ---- pipeline ----
  transpose_xyz_kernel<<<ceil_div(B * N0, 256), 256, 0, stream>>>(xyz_in, pts, B, N0);
  cvt_pad_kernel<<<ceil_div(B * N0 * 32, 256), 256, 0, stream>>>(pts, l0h, B * N0, 3, 32, B * N0);

  run_sa(pts, l0h, B, N0, 3, 32, 1024, 0.2f, 32, sa1, d_sa1, 3, l1_xyz, l1_pts, sc, stream);
  run_invres(l1_xyz, l1_pts, B, 1024, 64, 0.2f, 32, ir1, d_ir1, sc, stream);

  run_sa(l1_xyz, l1_pts, B, 1024, 64, 64, 256, 0.4f, 32, sa2, d_sa2, 3, l2_xyz, l2_pts, sc, stream);
  run_invres(l2_xyz, l2_pts, B, 256, 128, 0.4f, 32, ir2, d_ir2, sc, stream);

  run_sa(l2_xyz, l2_pts, B, 256, 128, 128, 64, 0.8f, 32, sa3, d_sa3, 3, l3_xyz, l3_pts, sc, stream);
  run_invres(l3_xyz, l3_pts, B, 64, 256, 0.8f, 32, ir3, d_ir3, sc, stream);

  run_sa(l3_xyz, l3_pts, B, 64, 256, 256, 16, 1.6f, 32, sa4, d_sa4, 3, l4_xyz, l4_pts, sc, stream);
  run_invres(l4_xyz, l4_pts, B, 16, 512, 1.6f, 16, ir4, d_ir4, sc, stream);

  run_fp(l3_xyz, l4_xyz, l3_pts, 256, l4_pts, 512, B, 64, 16,    fp4, d_fp4, 2, l3p, sc, stream);
  run_fp(l2_xyz, l3_xyz, l2_pts, 128, l3p,    256, B, 256, 64,   fp3, d_fp3, 2, l2p, sc, stream);
  run_fp(l1_xyz, l2_xyz, l1_pts, 64,  l2p,    256, B, 1024, 256, fp2, d_fp2, 2, l1p, sc, stream);
  run_fp(pts,    l1_xyz, nullptr, 0,  l1p,    128, B, N0, 1024,  fp1, d_fp1, 3, l0p, sc, stream);

  // head: bn+relu 128->128, plain 128->13 (padded to 32), log_softmax + transpose
  launch_linear(l0p, head1[0], sc.bufA, B * N0, 128, 128, 3, sc, stream);
  launch_linear(sc.bufA, head2, sc.bufB, B * N0, 128, 13, 0, sc, stream);
  head_logsoftmax_kernel<<<ceil_div(B * N0, 128), 128, 0, stream>>>(sc.bufB, (float*)d_out, B, N0, 13, 32);
}
